// PoincareEmbedding_72138270704140
// MI455X (gfx1250) — compile-verified, matrix operand order checked by reading, not run on hardware
//
#include <hip/hip_runtime.h>
#include <hip/hip_bf16.h>

typedef __attribute__((ext_vector_type(16))) __bf16 v16bf;
typedef __attribute__((ext_vector_type(8)))  __bf16 v8bf;
typedef __attribute__((ext_vector_type(8)))  float  v8f;
typedef __attribute__((ext_vector_type(4)))  unsigned int v4u;
typedef __attribute__((ext_vector_type(8)))  int v8i;
typedef __attribute__((ext_vector_type(4)))  int v4i;

#define EMBED_DIM   256
#define DEGREE      3
#define NPAIRS      (1024 * 50)
#define PAIRS_PER_WAVE 16
#define WAVES_PER_BLOCK 8
#define KTILES      8              // 256 / 32
#define MTILES      16             // 256 / 16
// packed M: [mt (16)][kc (8)][lane (32)][16 bf16] = 65536 bf16 = 128 KB
#define MPACK_ELEMS (MTILES * KTILES * 32 * 16)
#define MPACK_DWORDS (MPACK_ELEMS / 2)                   // 32768 x 4B

// ---------------------------------------------------------------------------
// Kernel 1: materialize the subtree mask M[j,i] (1 iff j is ancestor-or-self
// of i in the degree-3 heap tree), pre-swizzled into the CDNA5 16-bit
// A-fragment layout for v_wmma_f32_16x16x32_bf16 (ISA 7.12.2).
// ---------------------------------------------------------------------------
__global__ void build_m_packed_kernel(unsigned short* __restrict__ mp) {
    int t = blockIdx.x * blockDim.x + threadIdx.x;   // 0..65535
    int h    = t & 15;
    int lane = (t >> 4) & 31;
    int kc   = (t >> 9) & 7;
    int mt   = (t >> 12) & 15;

    int row   = mt * 16 + (lane & 15);                       // subtree root j
    int kbase = kc * 32 + ((lane & 16) ? 8 : 0);
    int col   = kbase + ((h < 8) ? h : (8 + h));             // h>=8 -> +16+(h-8)

    unsigned short v = 0;
    int j = col;                                             // walk col up to root
    for (;;) {
        if (j == row) { v = 0x3F80; break; }                 // bf16 1.0
        if (j == 0) break;
        j = (j - 1) / DEGREE;
    }
    mp[t] = v;
}

// ---------------------------------------------------------------------------
// Kernel 2: one wave32 per group of 16 (b,s) pairs.
//   TDM:     one tensor_load_to_lds per block stages the 128 KB mask into LDS,
//            overlapping with the HBM gather of phase 1.
//   Phase 1: gather + softmax-diff -> bf16 into wave-private LDS slab
//   Phase 2: z = M * D via v_wmma_f32_16x16x32_bf16 (A from LDS), sum |z|
// ---------------------------------------------------------------------------
__global__ __launch_bounds__(WAVES_PER_BLOCK * 32, 1)
void poincare_tree_l1_kernel(const float* __restrict__ weight,
                             const int*  __restrict__ xi,
                             const int*  __restrict__ yi,
                             const unsigned short* __restrict__ mpk,
                             float* __restrict__ out) {
    __shared__ unsigned short m_lds[MPACK_ELEMS];                        // 128 KB
    __shared__ __bf16 pd[WAVES_PER_BLOCK * PAIRS_PER_WAVE * EMBED_DIM]; //  64 KB

    const int lane  = threadIdx.x & 31;
    const int wave  = threadIdx.x >> 5;
    const int group = blockIdx.x * WAVES_PER_BLOCK + wave;   // 0..3199
    const int pbase = group * PAIRS_PER_WAVE;
    __bf16* mypd = &pd[wave * PAIRS_PER_WAVE * EMBED_DIM];

    // ---- TDM: async-copy packed M (global, L2-resident) -> LDS ------------
    if (wave == 0) {
        // Generic LDS pointers carry the LDS byte offset in addr[31:0]
        // (flat aperture mapping, ISA 10.2).
        unsigned lds_base = (unsigned)(unsigned long long)(void*)m_lds;
        unsigned long long ga = (unsigned long long)mpk;
        // D# group 0: count=1 | lds_addr | global_addr[56:0] | type=2
        v4u g0 = { 1u, lds_base, (unsigned)ga,
                   0x80000000u | (unsigned)(ga >> 32) };
        // D# group 1: 1-D tile, data_size=4B, dim0 = tile0 = 32768 dwords
        v8i g1 = { (int)(2u << 16),      // [17:16] data_size = 2 (4 bytes)
                   (int)0x80000000u,     // [63:48] tensor_dim0.lo16 = 0x8000
                   (int)0x00010000u,     // [95:80] tensor_dim1.lo16 = 1
                   (int)0x80000000u,     // [127:112] tile_dim0 = 0x8000
                   (int)0x00000001,      // [143:128] tile_dim1 = 1
                   (int)MPACK_DWORDS,    // [191:160] tensor_dim0_stride.lo
                   0, 0 };
        v4i g2 = {0, 0, 0, 0};
        v4i g3 = {0, 0, 0, 0};
        v8i g4 = {0, 0, 0, 0, 0, 0, 0, 0};
        __builtin_amdgcn_tensor_load_to_lds(g0, g1, g2, g3, g4, 0);
    }

    // ---------------- Phase 1: softmax diffs (overlaps the TDM copy) -------
    for (int j = 0; j < PAIRS_PER_WAVE; ++j) {
        const int p  = pbase + j;
        const int ix = xi[p];
        const int iy = yi[p];
        const float4* wx = (const float4*)(weight + (long)ix * EMBED_DIM) + lane * 2;
        const float4* wy = (const float4*)(weight + (long)iy * EMBED_DIM) + lane * 2;
        float4 a0 = wx[0], a1 = wx[1];
        float4 b0 = wy[0], b1 = wy[1];
        float ax[8] = {a0.x, a0.y, a0.z, a0.w, a1.x, a1.y, a1.z, a1.w};
        float ay[8] = {b0.x, b0.y, b0.z, b0.w, b1.x, b1.y, b1.z, b1.w};

        float mx = ax[0], my = ay[0];
#pragma unroll
        for (int e = 1; e < 8; ++e) { mx = fmaxf(mx, ax[e]); my = fmaxf(my, ay[e]); }
#pragma unroll
        for (int o = 16; o > 0; o >>= 1) {
            mx = fmaxf(mx, __shfl_xor(mx, o, 32));
            my = fmaxf(my, __shfl_xor(my, o, 32));
        }
        float sx = 0.f, sy = 0.f;
#pragma unroll
        for (int e = 0; e < 8; ++e) {
            ax[e] = __expf(ax[e] - mx); sx += ax[e];
            ay[e] = __expf(ay[e] - my); sy += ay[e];
        }
#pragma unroll
        for (int o = 16; o > 0; o >>= 1) {
            sx += __shfl_xor(sx, o, 32);
            sy += __shfl_xor(sy, o, 32);
        }
        const float rx = 1.f / sx, ry = 1.f / sy;
        v8bf d;
#pragma unroll
        for (int e = 0; e < 8; ++e) d[e] = (__bf16)(ax[e] * rx - ay[e] * ry);
        *(v8bf*)(mypd + j * EMBED_DIM + lane * 8) = d;       // ds_store_b128
    }

    // ---- B fragments (32x16 bf16) hoisted from the wave-private slab ------
    const int n    = lane & 15;
    const int koff = (lane & 16) ? 8 : 0;
    v16bf B[KTILES];
#pragma unroll
    for (int kc = 0; kc < KTILES; ++kc) {
        const __bf16* src = mypd + n * EMBED_DIM + kc * 32 + koff;
        v8bf lo = *(const v8bf*)(src);                       // ds_load_b128
        v8bf hi = *(const v8bf*)(src + 16);                  // ds_load_b128
        v16bf b;
#pragma unroll
        for (int e = 0; e < 8; ++e) { b[e] = lo[e]; b[8 + e] = hi[e]; }
        B[kc] = b;
    }

    // ---- Wait for the TDM copy, then make M visible to all waves ----------
    if (wave == 0) __builtin_amdgcn_s_wait_tensorcnt(0);
    __syncthreads();

    // ---------------- Phase 2: WMMA z = M * D, accumulate |z| --------------
    const v16bf* Alds = (const v16bf*)m_lds;
    float s = 0.f;
#pragma unroll 2
    for (int mt = 0; mt < MTILES; ++mt) {
        v8f c = {};
#pragma unroll
        for (int kc = 0; kc < KTILES; ++kc) {
            v16bf a = Alds[(mt * KTILES + kc) * 32 + lane];  // 2x ds_load_b128
            c = __builtin_amdgcn_wmma_f32_16x16x32_bf16(
                    /*neg_a=*/false, a, /*neg_b=*/false, B[kc],
                    /*c_mod=*/(short)0, c, /*reuse_a=*/false, /*reuse_b=*/false);
        }
#pragma unroll
        for (int v = 0; v < 8; ++v) s += fabsf(c[v]);
    }
    // column n total = lane n (rows 0-7 of each tile) + lane n+16 (rows 8-15)
    s += __shfl_xor(s, 16, 32);
    if (lane < 16) out[pbase + lane] = s;
}

// ---------------------------------------------------------------------------
extern "C" void kernel_launch(void* const* d_in, const int* in_sizes, int n_in,
                              void* d_out, int out_size, void* d_ws, size_t ws_size,
                              hipStream_t stream) {
    const float* weight = (const float*)d_in[0];
    const int*   x      = (const int*)d_in[1];
    const int*   y      = (const int*)d_in[2];
    float*       out    = (float*)d_out;
    unsigned short* mpk = (unsigned short*)d_ws;             // 128 KB packed mask

    build_m_packed_kernel<<<MPACK_ELEMS / 256, 256, 0, stream>>>(mpk);

    const int groups = NPAIRS / PAIRS_PER_WAVE;              // 3200 waves
    const int blocks = groups / WAVES_PER_BLOCK;             // 400 blocks
    poincare_tree_l1_kernel<<<blocks, WAVES_PER_BLOCK * 32, 0, stream>>>(
        weight, x, y, mpk, out);
}
